// SkillVAE_57638461112646
// MI455X (gfx1250) — matrix operand order repaired and checked by reading, not compiled, not run on hardware
//
#include <hip/hip_runtime.h>
#include <hip/hip_bf16.h>

typedef __attribute__((ext_vector_type(16))) __bf16 v16bf;
typedef __attribute__((ext_vector_type(8)))  float  v8f;
typedef __attribute__((ext_vector_type(4)))  __bf16 bf16x4;
typedef __attribute__((ext_vector_type(4)))  float  float4v;

#define TM 64
#define TN 32
#define TK 32
#define LDA 36   // bf16 elements per LDS row (72B: 8B-aligned, padded)

// ---------------------------------------------------------------------------
// WMMA: one wave computes a 16x32 f32 tile (two 16x16 accumulators, A reused).
// Fragments: 16-bit A 16x32, B 32x16 with contiguous k = 16*half + e in both; f32 C/D.
// ---------------------------------------------------------------------------
__device__ __forceinline__ void wmma_tile2(const __bf16* As, const __bf16* Ws,
                                           v8f& acc0, v8f& acc1) {
  int lane = threadIdx.x & 31;
  int half = lane >> 4;
  int r    = lane & 15;
  v16bf a, b0, b1;
#pragma unroll
  for (int e = 0; e < 16; ++e) {
    int k = half * 16 + e;
    a[e] = As[r * LDA + k];
  }
#pragma unroll
  for (int e = 0; e < 16; ++e) {
    int k = half * 16 + e;
    b0[e] = Ws[r * LDA + k];
    b1[e] = Ws[(16 + r) * LDA + k];
  }
  acc0 = __builtin_amdgcn_wmma_f32_16x16x32_bf16(false, a, false, b0, (short)0, acc0,
                                                 false, false);
  acc1 = __builtin_amdgcn_wmma_f32_16x16x32_bf16(false, a, false, b1, (short)0, acc1,
                                                 false, false);
}

// ---------------------------------------------------------------------------
// Generic GEMM: out[M,N] = act(A[M,K] * W[N,K]^T + bias + bias2 + addsrc)
// 128 threads (4 waves), block tile 64x32, TK=32 bf16 chunks,
// LDS double-buffered + software pipelined (loads for k+1 overlap WMMA on k).
// Contract: M % 64 == 0. kAligned => K % 32 == 0 (vectorized b128 staging).
// ---------------------------------------------------------------------------
__global__ __launch_bounds__(128)
void gemm_bias_act(const float* __restrict__ A, const float* __restrict__ W,
                   const float* __restrict__ bias, const float* __restrict__ bias2,
                   const float* addsrc, float* out,
                   int M, int N, int K, int act, int kAligned) {
  __shared__ __bf16 As[2][TM][LDA];
  __shared__ __bf16 Ws[2][TN][LDA];
  int tid  = threadIdx.x;
  int wave = tid >> 5;
  int m0 = blockIdx.x * TM;
  int n0 = blockIdx.y * TN;
  v8f acc0 = {}, acc1 = {};

  float4v ra[4], rw[2];

  // phase 1: issue all global loads for chunk at k0 (no waits interleaved)
  auto load_chunk = [&](int k0) {
    if (kAligned) {
#pragma unroll
      for (int i = 0; i < 4; ++i) {
        int idx = tid + i * 128;
        int r = idx >> 3, c4 = idx & 7;
        ra[i] = *(const float4v*)(A + (size_t)(m0 + r) * K + k0 + c4 * 4);
      }
#pragma unroll
      for (int i = 0; i < 2; ++i) {
        int idx = tid + i * 128;
        int r = idx >> 3, c4 = idx & 7;
        int gn = n0 + r;
        int rn = (gn < N) ? gn : (N - 1);
        rw[i] = *(const float4v*)(W + (size_t)rn * K + k0 + c4 * 4);
      }
    } else {
#pragma unroll
      for (int i = 0; i < 16; ++i) {
        int idx = tid + i * 128;
        int r = idx >> 5, c = idx & 31;
        int gk = k0 + c;
        int kk = (gk < K) ? gk : (K - 1);
        ra[i >> 2][i & 3] = A[(size_t)(m0 + r) * K + kk];
      }
#pragma unroll
      for (int i = 0; i < 8; ++i) {
        int idx = tid + i * 128;
        int r = idx >> 5, c = idx & 31;
        int gn = n0 + r, gk = k0 + c;
        int rn = (gn < N) ? gn : (N - 1);
        int kk = (gk < K) ? gk : (K - 1);
        rw[i >> 2][i & 3] = W[(size_t)rn * K + kk];
      }
    }
  };

  // phase 2: convert + store registers into LDS buffer `buf`
  auto store_chunk = [&](int buf, int k0) {
    if (kAligned) {
#pragma unroll
      for (int i = 0; i < 4; ++i) {
        int idx = tid + i * 128;
        int r = idx >> 3, c4 = idx & 7;
        bf16x4 q;
#pragma unroll
        for (int j = 0; j < 4; ++j) q[j] = (__bf16)ra[i][j];
        *(bf16x4*)&As[buf][r][c4 * 4] = q;
      }
#pragma unroll
      for (int i = 0; i < 2; ++i) {
        int idx = tid + i * 128;
        int r = idx >> 3, c4 = idx & 7;
        int gn = n0 + r;
        bf16x4 q;
#pragma unroll
        for (int j = 0; j < 4; ++j) q[j] = (__bf16)((gn < N) ? rw[i][j] : 0.f);
        *(bf16x4*)&Ws[buf][r][c4 * 4] = q;
      }
    } else {
#pragma unroll
      for (int i = 0; i < 16; ++i) {
        int idx = tid + i * 128;
        int r = idx >> 5, c = idx & 31;
        int gk = k0 + c;
        As[buf][r][c] = (__bf16)((gk < K) ? ra[i >> 2][i & 3] : 0.f);
      }
#pragma unroll
      for (int i = 0; i < 8; ++i) {
        int idx = tid + i * 128;
        int r = idx >> 5, c = idx & 31;
        int gn = n0 + r, gk = k0 + c;
        Ws[buf][r][c] = (__bf16)((gn < N && gk < K) ? rw[i >> 2][i & 3] : 0.f);
      }
    }
  };

  int nk = (K + TK - 1) / TK;
  load_chunk(0);
  store_chunk(0, 0);
  __syncthreads();
  for (int kc = 0; kc < nk; ++kc) {
    int cur = kc & 1;
    bool have_next = (kc + 1) < nk;
    if (have_next) load_chunk((kc + 1) * TK);          // global loads in flight
    wmma_tile2(&As[cur][wave * 16][0], &Ws[cur][0][0], acc0, acc1);  // math on cur
    if (have_next) store_chunk(cur ^ 1, (kc + 1) * TK);// wait loads, fill other buf
    __syncthreads();
  }

  int lane = tid & 31, half = lane >> 4, r = lane & 15;
#pragma unroll
  for (int sub = 0; sub < 2; ++sub) {
    v8f& acc = sub ? acc1 : acc0;
#pragma unroll
    for (int i = 0; i < 8; ++i) {
      int row = m0 + wave * 16 + half * 8 + i;
      int col = n0 + sub * 16 + r;
      if (col < N) {
        float v = acc[i];
        if (bias)   v += bias[col];
        if (bias2)  v += bias2[col];
        if (addsrc) v += addsrc[(size_t)row * N + col];
        if (act == 1) v = fmaxf(v, 0.f);
        out[(size_t)row * N + col] = v;
      }
    }
  }
}

// ---------------------------------------------------------------------------
// 3x3 pad-1 conv on 5x13 maps as implicit-im2col WMMA GEMM.
// M = frames*65 (mult of 64), N = Cout, K = Cin*9. NCHW out, bias+ReLU fused.
// Same double-buffered pipeline; branch-free im2col gather (clamp + select).
// ---------------------------------------------------------------------------
__global__ __launch_bounds__(128)
void conv3x3_wmma(const float* __restrict__ in, const float* __restrict__ w,
                  const float* __restrict__ bias, float* __restrict__ out,
                  int frames, int Cin, int Cout) {
  const int HH = 5, WW = 13, PIX = 65;
  int K = Cin * 9;
  __shared__ __bf16 As[2][TM][LDA];
  __shared__ __bf16 Ws[2][TN][LDA];
  int tid  = threadIdx.x;
  int wave = tid >> 5;
  int m0 = blockIdx.x * TM;
  int n0 = blockIdx.y * TN;
  v8f acc0 = {}, acc1 = {};

  float ra[16]; float rw[8];

  auto load_chunk = [&](int k0) {
#pragma unroll
    for (int i = 0; i < 16; ++i) {
      int idx = tid + i * 128;
      int r = idx >> 5, c = idx & 31;
      int gm = m0 + r;
      int gk = k0 + c;
      int kk = (gk < K) ? gk : (K - 1);
      int f = gm / PIX, fpix = gm % PIX;
      int y = fpix / WW, x = fpix % WW;
      int ic = kk / 9, rr = kk - ic * 9;
      int ky = rr / 3, kx = rr - ky * 3;
      int iy = y + ky - 1, ix = x + kx - 1;
      int cy = min(max(iy, 0), HH - 1), cx = min(max(ix, 0), WW - 1);
      ra[i] = in[((size_t)f * Cin + ic) * PIX + cy * WW + cx];
    }
#pragma unroll
    for (int i = 0; i < 8; ++i) {
      int idx = tid + i * 128;
      int r = idx >> 5, c = idx & 31;
      int gn = n0 + r, gk = k0 + c;
      int rn = (gn < Cout) ? gn : (Cout - 1);
      int kk = (gk < K) ? gk : (K - 1);
      rw[i] = w[(size_t)rn * K + kk];
    }
  };

  auto store_chunk = [&](int buf, int k0) {
#pragma unroll
    for (int i = 0; i < 16; ++i) {
      int idx = tid + i * 128;
      int r = idx >> 5, c = idx & 31;
      int gm = m0 + r;
      int gk = k0 + c;
      int kk = (gk < K) ? gk : (K - 1);
      int fpix = gm % PIX;
      int y = fpix / WW, x = fpix % WW;
      int rr = kk - (kk / 9) * 9;
      int ky = rr / 3, kx = rr - ky * 3;
      int iy = y + ky - 1, ix = x + kx - 1;
      bool ok = (gk < K) & (iy >= 0) & (iy < HH) & (ix >= 0) & (ix < WW);
      As[buf][r][c] = (__bf16)(ok ? ra[i] : 0.f);
    }
#pragma unroll
    for (int i = 0; i < 8; ++i) {
      int idx = tid + i * 128;
      int r = idx >> 5, c = idx & 31;
      int gn = n0 + r, gk = k0 + c;
      Ws[buf][r][c] = (__bf16)((gn < Cout && gk < K) ? rw[i] : 0.f);
    }
  };

  int nk = (K + TK - 1) / TK;
  load_chunk(0);
  store_chunk(0, 0);
  __syncthreads();
  for (int kc = 0; kc < nk; ++kc) {
    int cur = kc & 1;
    bool have_next = (kc + 1) < nk;
    if (have_next) load_chunk((kc + 1) * TK);
    wmma_tile2(&As[cur][wave * 16][0], &Ws[cur][0][0], acc0, acc1);
    if (have_next) store_chunk(cur ^ 1, (kc + 1) * TK);
    __syncthreads();
  }

  int lane = tid & 31, half = lane >> 4, r = lane & 15;
#pragma unroll
  for (int sub = 0; sub < 2; ++sub) {
    v8f& acc = sub ? acc1 : acc0;
#pragma unroll
    for (int i = 0; i < 8; ++i) {
      int row = m0 + wave * 16 + half * 8 + i;
      int oc  = n0 + sub * 16 + r;
      if (oc < Cout) {
        int f = row / PIX, pix = row % PIX;
        float v = fmaxf(acc[i] + bias[oc], 0.f);
        out[((size_t)f * Cout + oc) * PIX + pix] = v;
      }
    }
  }
}

// ---------------------------------------------------------------------------
// LayerNorm (+ReLU) over cols, one block per row.
// ---------------------------------------------------------------------------
__global__ __launch_bounds__(256)
void layernorm_relu(const float* __restrict__ in, const float* __restrict__ g,
                    const float* __restrict__ b, float* __restrict__ out, int cols) {
  __shared__ float red[256];
  int row = blockIdx.x;
  const float* x = in + (size_t)row * cols;
  float s = 0.f;
  for (int j = threadIdx.x; j < cols; j += 256) s += x[j];
  red[threadIdx.x] = s; __syncthreads();
  for (int off = 128; off > 0; off >>= 1) {
    if (threadIdx.x < off) red[threadIdx.x] += red[threadIdx.x + off];
    __syncthreads();
  }
  float mean = red[0] / cols; __syncthreads();
  float v = 0.f;
  for (int j = threadIdx.x; j < cols; j += 256) { float d = x[j] - mean; v += d * d; }
  red[threadIdx.x] = v; __syncthreads();
  for (int off = 128; off > 0; off >>= 1) {
    if (threadIdx.x < off) red[threadIdx.x] += red[threadIdx.x + off];
    __syncthreads();
  }
  float inv = rsqrtf(red[0] / cols + 1e-5f);
  for (int j = threadIdx.x; j < cols; j += 256) {
    float y = (x[j] - mean) * inv * g[j] + b[j];
    out[(size_t)row * cols + j] = fmaxf(y, 0.f);
  }
}

// ---------------------------------------------------------------------------
// LSTM cell elementwise (gates already = x·Wih^T + h·Whh^T + bih + bhh).
// ---------------------------------------------------------------------------
__global__ void lstm_cell(const float* __restrict__ g, float* __restrict__ h,
                          float* __restrict__ c, float* __restrict__ dst,
                          int B, int dstride, int doff) {
  int idx = blockIdx.x * blockDim.x + threadIdx.x;
  if (idx >= B * 128) return;
  int b = idx >> 7, j = idx & 127;
  const float* gr = g + (size_t)b * 512;
  float gi = gr[j], gf = gr[128 + j], gg = gr[256 + j], go = gr[384 + j];
  float si = 1.f / (1.f + expf(-gi));
  float sf = 1.f / (1.f + expf(-gf));
  float so = 1.f / (1.f + expf(-go));
  float cn = sf * c[idx] + si * tanhf(gg);
  float hn = so * tanhf(cn);
  c[idx] = cn; h[idx] = hn;
  dst[(size_t)b * dstride + doff + j] = hn;
}

__global__ void fill_kernel(float* p, float v, long n) {
  long i = (long)blockIdx.x * blockDim.x + threadIdx.x;
  if (i < n) p[i] = v;
}

// xs0[t][b][0:256]=cnn[(b*5+t)], [256:288]=aemb[(b*5+t)], [288:320]=temb[b]
__global__ void build_xs0(const float* __restrict__ cnn, const float* __restrict__ ae,
                          const float* __restrict__ te, float* __restrict__ xs) {
  long i = (long)blockIdx.x * blockDim.x + threadIdx.x;
  if (i >= 5L * 2048 * 320) return;
  int t = (int)(i / (2048L * 320));
  int rem = (int)(i % (2048L * 320));
  int b = rem / 320, j = rem % 320;
  float v;
  if (j < 256)      v = cnn[(size_t)(b * 5 + t) * 256 + j];
  else if (j < 288) v = ae[(size_t)(b * 5 + t) * 32 + (j - 256)];
  else              v = te[(size_t)b * 32 + (j - 288)];
  xs[i] = v;
}

__global__ void pool_mean_t(const float* __restrict__ xs, float* __restrict__ out) {
  int i = blockIdx.x * blockDim.x + threadIdx.x;
  if (i >= 2048 * 256) return;
  float s = 0.f;
  for (int t = 0; t < 5; ++t) s += xs[(size_t)t * 2048 * 256 + i];
  out[i] = s * 0.2f;
}

__global__ void norm_rows(const float* __restrict__ in, float* __restrict__ out,
                          int rows, int cols) {
  int r = blockIdx.x * blockDim.x + threadIdx.x;
  if (r >= rows) return;
  float s = 0.f;
  for (int j = 0; j < cols; ++j) { float v = in[(size_t)r * cols + j]; s += v * v; }
  float d = 1.f / fmaxf(sqrtf(s), 1e-12f);
  for (int j = 0; j < cols; ++j) out[(size_t)r * cols + j] = in[(size_t)r * cols + j] * d;
}

__global__ void vq_assign(const float* __restrict__ z, const float* __restrict__ e,
                          float* __restrict__ zq, float* __restrict__ idx_out,
                          float* __restrict__ lossacc, float* __restrict__ counts, int B) {
  int b = blockIdx.x * blockDim.x + threadIdx.x;
  if (b >= B) return;
  const float* zr = z + (size_t)b * 16;
  float best = 3.4e38f; int bi = 0;
  for (int k = 0; k < 128; ++k) {
    const float* er = e + k * 16;
    float zz = 0.f, ee = 0.f, ze = 0.f;
#pragma unroll
    for (int j = 0; j < 16; ++j) { zz += zr[j] * zr[j]; ee += er[j] * er[j]; ze += zr[j] * er[j]; }
    float d = zz + ee - 2.f * ze;
    if (d < best) { best = d; bi = k; }
  }
  idx_out[b] = (float)bi;
  float se = 0.f;
#pragma unroll
  for (int j = 0; j < 16; ++j) {
    float q = e[bi * 16 + j];
    zq[(size_t)b * 16 + j] = q;        // z_q_st forward value == z_q
    float df = q - zr[j]; se += df * df;
  }
  atomicAdd(lossacc, se);
  atomicAdd(&counts[bi], 1.f);
}

__global__ void vq_finalize(const float* __restrict__ lossacc, const float* __restrict__ counts,
                            float* vql, float* cbl, float* cml, float* perp, int B) {
  if (threadIdx.x != 0 || blockIdx.x != 0) return;
  float mse = lossacc[0] / (float)(B * 16);
  cbl[0] = mse; cml[0] = mse; vql[0] = mse + 0.25f * mse;
  float s = 0.f;
  for (int k = 0; k < 128; ++k) { float p = counts[k] / (float)B; s += p * logf(p + 1e-10f); }
  perp[0] = expf(-s);
}

__global__ void build_dec_in(const float* __restrict__ zq, const float* __restrict__ task,
                             float* __restrict__ out) {
  int i = blockIdx.x * blockDim.x + threadIdx.x;
  if (i >= 2048 * 26) return;
  int b = i / 26, j = i % 26;
  out[i] = (j < 16) ? zq[(size_t)b * 16 + j] : task[(size_t)b * 10 + (j - 16)];
}

// ---------------------------------------------------------------------------
// Host orchestration
// ---------------------------------------------------------------------------
static inline dim3 gemm_grid(int M, int N) { return dim3(M / TM, (N + TN - 1) / TN); }

extern "C" void kernel_launch(void* const* d_in, const int* in_sizes, int n_in,
                              void* d_out, int out_size, void* d_ws, size_t ws_size,
                              hipStream_t stream) {
  const int B = 2048, T = 5, FT = B * T;           // 10240 frames
  const float* obs    = (const float*)d_in[0];
  const float* action = (const float*)d_in[1];
  const float* task   = (const float*)d_in[2];
  const float* conv1w = (const float*)d_in[3];  const float* conv1b = (const float*)d_in[4];
  const float* conv2w = (const float*)d_in[5];  const float* conv2b = (const float*)d_in[6];
  const float* fcw    = (const float*)d_in[7];  const float* fcb    = (const float*)d_in[8];
  const float* lng    = (const float*)d_in[9];  const float* lnb    = (const float*)d_in[10];
  const float* aembw  = (const float*)d_in[11]; const float* aembb  = (const float*)d_in[12];
  const float* tembw  = (const float*)d_in[13]; const float* tembb  = (const float*)d_in[14];
  const float* fczw   = (const float*)d_in[15]; const float* fczb   = (const float*)d_in[16];
  const float* emb    = (const float*)d_in[17];
  const float* dec1w  = (const float*)d_in[18]; const float* dec1b  = (const float*)d_in[19];
  const float* dec2w  = (const float*)d_in[20]; const float* dec2b  = (const float*)d_in[21];
  const float* fhw    = (const float*)d_in[22]; const float* fhb    = (const float*)d_in[23];
  const float* ahw    = (const float*)d_in[24]; const float* ahb    = (const float*)d_in[25];
  // lstm: l0f=26..29, l0b=30..33, l1f=34..37, l1b=38..41 (Wih,Whh,bih,bhh)

  float* out = (float*)d_out;
  float* o_ze   = out;                         // 2048*16
  float* o_zq   = out + 32768;                 // 2048*16
  float* o_pf   = out + 65536;                 // 2048*5*256
  float* o_pa   = out + 2686976;               // 2048*5*6
  float* o_cnn  = out + 2748416;               // 2048*5*256
  float* o_vql  = out + 5369856;
  float* o_cbl  = out + 5369857;
  float* o_cml  = out + 5369858;
  float* o_perp = out + 5369859;
  float* o_idx  = out + 5369860;               // 2048

  float* ws = (float*)d_ws;
  size_t o = 0;
  float* x1     = ws + o; o += (size_t)FT * 32 * 65;
  float* x2     = ws + o; o += (size_t)FT * 64 * 65;
  float* fcbuf  = ws + o; o += (size_t)FT * 256;
  float* aebuf  = ws + o; o += (size_t)FT * 32;
  float* tebuf  = ws + o; o += (size_t)B * 32;
  float* xs0    = ws + o; o += (size_t)T * B * 320;
  float* xs1    = ws + o; o += (size_t)T * B * 256;
  float* xs2    = ws + o; o += (size_t)T * B * 256;
  float* gbuf   = ws + o; o += (size_t)FT * 512;
  float* hbuf   = ws + o; o += (size_t)B * 128;
  float* cbuf   = ws + o; o += (size_t)B * 128;
  float* pooled = ws + o; o += (size_t)B * 256;
  float* flatz  = ws + o; o += (size_t)B * 16;
  float* embn   = ws + o; o += (size_t)128 * 16;
  float* vqacc  = ws + o; o += 1 + 128;        // [0]=loss sum, [1..128]=counts
  float* decin  = ws + o; o += (size_t)B * 26;
  float* h1     = ws + o; o += (size_t)B * 256;
  float* h2     = ws + o; o += (size_t)B * 256;
  (void)ws_size; (void)n_in; (void)in_sizes; (void)out_size;

  // ---- encoder convs (implicit-im2col WMMA) ----
  conv3x3_wmma<<<dim3(FT * 65 / TM, 1), 128, 0, stream>>>(obs, conv1w, conv1b, x1, FT, 26, 32);
  conv3x3_wmma<<<dim3(FT * 65 / TM, 2), 128, 0, stream>>>(x1, conv2w, conv2b, x2, FT, 32, 64);
  // ---- fc + layernorm/relu -> cnn_features (also an output) ----
  gemm_bias_act<<<gemm_grid(FT, 256), 128, 0, stream>>>(x2, fcw, fcb, nullptr, nullptr, fcbuf, FT, 256, 4160, 0, 1);
  layernorm_relu<<<FT, 256, 0, stream>>>(fcbuf, lng, lnb, o_cnn, 256);
  // ---- embeddings ----
  gemm_bias_act<<<gemm_grid(FT, 32), 128, 0, stream>>>(action, aembw, aembb, nullptr, nullptr, aebuf, FT, 32, 6, 0, 0);
  gemm_bias_act<<<gemm_grid(B, 32), 128, 0, stream>>>(task, tembw, tembb, nullptr, nullptr, tebuf, B, 32, 10, 0, 0);
  build_xs0<<<(5L * 2048 * 320 + 255) / 256, 256, 0, stream>>>(o_cnn, aebuf, tebuf, xs0);

  // ---- 2-layer bidirectional LSTM ----
  for (int layer = 0; layer < 2; ++layer) {
    const float* xin = (layer == 0) ? xs0 : xs1;
    float* xout = (layer == 0) ? xs1 : xs2;
    int K = (layer == 0) ? 320 : 256;
    int base = 26 + layer * 8;
    for (int dir = 0; dir < 2; ++dir) {
      const float* Wih = (const float*)d_in[base + dir * 4 + 0];
      const float* Whh = (const float*)d_in[base + dir * 4 + 1];
      const float* bih = (const float*)d_in[base + dir * 4 + 2];
      const float* bhh = (const float*)d_in[base + dir * 4 + 3];
      gemm_bias_act<<<gemm_grid(FT, 512), 128, 0, stream>>>(xin, Wih, bih, bhh, nullptr, gbuf, FT, 512, K, 0, 1);
      fill_kernel<<<(B * 128 + 255) / 256, 256, 0, stream>>>(hbuf, 0.f, B * 128);
      fill_kernel<<<(B * 128 + 255) / 256, 256, 0, stream>>>(cbuf, 0.f, B * 128);
      for (int s = 0; s < T; ++s) {
        int t = dir ? (T - 1 - s) : s;
        float* gt = gbuf + (size_t)t * B * 512;
        gemm_bias_act<<<gemm_grid(B, 512), 128, 0, stream>>>(hbuf, Whh, nullptr, nullptr, gt, gt, B, 512, 128, 0, 1);
        lstm_cell<<<(B * 128 + 255) / 256, 256, 0, stream>>>(gt, hbuf, cbuf,
            xout + (size_t)t * B * 256, B, 256, dir * 128);
      }
    }
  }
  pool_mean_t<<<(2048 * 256 + 255) / 256, 256, 0, stream>>>(xs2, pooled);
  // ---- z_e ----
  gemm_bias_act<<<gemm_grid(B, 16), 128, 0, stream>>>(pooled, fczw, fczb, nullptr, nullptr, o_ze, B, 16, 256, 0, 1);

  // ---- vector quantizer ----
  norm_rows<<<(B + 127) / 128, 128, 0, stream>>>(o_ze, flatz, B, 16);
  norm_rows<<<1, 128, 0, stream>>>(emb, embn, 128, 16);
  fill_kernel<<<1, 256, 0, stream>>>(vqacc, 0.f, 129);
  vq_assign<<<(B + 127) / 128, 128, 0, stream>>>(flatz, embn, o_zq, o_idx, vqacc, vqacc + 1, B);
  vq_finalize<<<1, 32, 0, stream>>>(vqacc, vqacc + 1, o_vql, o_cbl, o_cml, o_perp, B);

  // ---- decoder ----
  build_dec_in<<<(2048 * 26 + 255) / 256, 256, 0, stream>>>(o_zq, task, decin);
  gemm_bias_act<<<gemm_grid(B, 256), 128, 0, stream>>>(decin, dec1w, dec1b, nullptr, nullptr, h1, B, 256, 26, 1, 0);
  gemm_bias_act<<<gemm_grid(B, 256), 128, 0, stream>>>(h1, dec2w, dec2b, nullptr, nullptr, h2, B, 256, 256, 1, 1);
  gemm_bias_act<<<gemm_grid(B, 1280), 128, 0, stream>>>(h2, fhw, fhb, nullptr, nullptr, o_pf, B, 1280, 256, 0, 1);
  gemm_bias_act<<<gemm_grid(B, 30), 128, 0, stream>>>(h2, ahw, ahb, nullptr, nullptr, o_pa, B, 30, 256, 0, 1);
}
